// LNSLinear_4475355922620
// MI455X (gfx1250) — matrix operand order repaired and checked by reading, not compiled, not run on hardware
//
#include <hip/hip_runtime.h>

typedef __attribute__((ext_vector_type(16))) __bf16  v16bf;
typedef __attribute__((ext_vector_type(8)))  float   v8f;
typedef __attribute__((ext_vector_type(4)))  float   v4f;

#define LNS_EPS 1e-8f
#define KD 256
#define ND 256
#define BSTR 264            // padded LDS row stride (elements): conflict-free ds_load_b128
#define LDS_BYTES (2 * 128 * BSTR * 2)   // two planes * 128 rows * BSTR * sizeof(bf16)

// 32-byte chunk with only 16-byte alignment (LDS rows are 16B-aligned after padding).
struct ChunkB32 { uint4 a, b; };

// sign(v) as a float in {-1, 0, +1}: one mul + one med3 (clamp).
static __device__ __forceinline__ float sgnf_of(float v) {
  return __builtin_amdgcn_fmed3f(v * 1e38f, -1.0f, 1.0f);
}

static __device__ __forceinline__ v16bf lds_frag(const __bf16* p) {
  return __builtin_bit_cast(v16bf, *(const ChunkB32*)p);
}

// ---------------------------------------------------------------------------
// Pre-pass: weight [N,K] f32 -> bf16 magnitudes (|w|+eps) and bf16 signs.
// ---------------------------------------------------------------------------
__global__ __launch_bounds__(256) void lns_cvt_weight(
    const float* __restrict__ w, __bf16* __restrict__ wbm, __bf16* __restrict__ wbs) {
  int i = blockIdx.x * blockDim.x + threadIdx.x;
  float v = w[i];
  wbm[i] = (__bf16)(fabsf(v) + LNS_EPS);
  wbs[i] = (__bf16)sgnf_of(v);     // exact: -1, 0, +1
}

// ---------------------------------------------------------------------------
// out[m,n] = sign(sum_k sgn(x)sgn(w)) * sum_k |x||w| + bias[n]
// Wave tile: 16(M) x 128(N). Workgroup: 8 waves = 128(M) x 128(N).
// B (mag+sign planes, 128 KB) staged once per workgroup in LDS.
// ---------------------------------------------------------------------------
__global__ __launch_bounds__(256) void lns_gemm(
    const float* __restrict__ x, const __bf16* __restrict__ wbm,
    const __bf16* __restrict__ wbs, const float* __restrict__ bias,
    float* __restrict__ out) {
  extern __shared__ __bf16 smem[];
  __bf16* sbm = smem;                    // [128][BSTR] magnitudes
  __bf16* sbs = smem + 128 * BSTR;       // [128][BSTR] signs

  const int lane   = threadIdx.x & 31;
  const int wave   = threadIdx.x >> 5;
  const int laneLo = lane & 15;
  const int laneHi = lane >> 4;          // 0 or 1

  const long m0    = (long)blockIdx.y * 128 + (long)wave * 16;
  const int  nbase = blockIdx.x * 128;

  // ---- cooperative stage of this block's weight slice into LDS ----
  {
    const int tid  = threadIdx.x;
    const int row  = tid >> 1;           // 0..127
    const int half = tid & 1;            // half-row of 128 elements (256 B)
    const uint4* gm = (const uint4*)(wbm + (size_t)(nbase + row) * KD + half * 128);
    const uint4* gs = (const uint4*)(wbs + (size_t)(nbase + row) * KD + half * 128);
    uint4* dm = (uint4*)(sbm + row * BSTR + half * 128);
    uint4* ds = (uint4*)(sbs + row * BSTR + half * 128);
#pragma unroll
    for (int i = 0; i < 16; ++i) { dm[i] = gm[i]; }
#pragma unroll
    for (int i = 0; i < 16; ++i) { ds[i] = gs[i]; }
  }
  __syncthreads();

  const float* xrow = x + (m0 + laneLo) * KD;

  v8f accf[8];   // magnitude accumulators
  v8f accs[8];   // sign accumulators (exact small integers in f32)
#pragma unroll
  for (int t = 0; t < 8; ++t)
#pragma unroll
    for (int j = 0; j < 8; ++j) { accf[t][j] = 0.0f; accs[t][j] = 0.0f; }

  for (int kk = 0; kk < KD; kk += 64) {
    // Per-lane x loads: 4 groups of 8 contiguous f32 at K = kk + laneHi*16 + {0,8,32,40}
    // = the union of two bf16 16x32 A-fragments (k-bases kk, kk+32).
    float a[32];
    const float* p = xrow + kk + laneHi * 16;
#pragma unroll
    for (int g = 0; g < 4; ++g) {
      v4f u0 = *(const v4f*)(p + (g >> 1) * 32 + (g & 1) * 8);
      v4f u1 = *(const v4f*)(p + (g >> 1) * 32 + (g & 1) * 8 + 4);
#pragma unroll
      for (int j = 0; j < 4; ++j) { a[g * 8 + j] = u0[j]; a[g * 8 + 4 + j] = u1[j]; }
    }

    v16bf am0, am1, as0, as1;
#pragma unroll
    for (int j = 0; j < 16; ++j) {
      am0[j] = (__bf16)fabsf(a[j]);          // abs folds into cvt src modifier
      am1[j] = (__bf16)fabsf(a[16 + j]);
      as0[j] = (__bf16)sgnf_of(a[j]);        // exact -1/0/+1
      as1[j] = (__bf16)sgnf_of(a[16 + j]);
    }

#pragma unroll
    for (int t = 0; t < 8; ++t) {
      const int boff = (t * 16 + laneLo) * BSTR + kk + laneHi * 16;
      v16bf bm0 = lds_frag(sbm + boff);        // K = kk    + laneHi*16 .. +15
      v16bf bm1 = lds_frag(sbm + boff + 32);   // K = kk+32 + laneHi*16 .. +15
      v16bf bs0 = lds_frag(sbs + boff);
      v16bf bs1 = lds_frag(sbs + boff + 32);

      accf[t] = __builtin_amdgcn_wmma_f32_16x16x32_bf16(
          false, am0, false, bm0, (short)0, accf[t], false, false);
      accf[t] = __builtin_amdgcn_wmma_f32_16x16x32_bf16(
          false, am1, false, bm1, (short)0, accf[t], false, false);
      accs[t] = __builtin_amdgcn_wmma_f32_16x16x32_bf16(
          false, as0, false, bs0, (short)0, accs[t], false, false);
      accs[t] = __builtin_amdgcn_wmma_f32_16x16x32_bf16(
          false, as1, false, bs1, (short)0, accs[t], false, false);
    }
  }

  // Epilogue: element v of a tile is (m0 + v + laneHi*8, laneLo). accs is an
  // exact integer-valued float, so med3(accs,-1,1) is its sign. 2 VALU/output.
  const long row0 = m0 + laneHi * 8;
#pragma unroll
  for (int t = 0; t < 8; ++t) {
    const int col = nbase + t * 16 + laneLo;
    const float bv = bias[col];
#pragma unroll
    for (int v = 0; v < 8; ++v) {
      const float sgn = __builtin_amdgcn_fmed3f(accs[t][v], -1.0f, 1.0f);
      out[(row0 + v) * ND + col] = fmaf(sgn, accf[t][v], bv);
    }
  }
}

extern "C" void kernel_launch(void* const* d_in, const int* in_sizes, int n_in,
                              void* d_out, int out_size, void* d_ws, size_t ws_size,
                              hipStream_t stream) {
  (void)n_in; (void)out_size; (void)ws_size;
  const float* x    = (const float*)d_in[0];
  const float* w    = (const float*)d_in[1];
  const float* bias = (const float*)d_in[2];
  float* out = (float*)d_out;

  __bf16* wbm = (__bf16*)d_ws;
  __bf16* wbs = (__bf16*)((char*)d_ws + (size_t)ND * KD * sizeof(__bf16));

  lns_cvt_weight<<<(ND * KD) / 256, 256, 0, stream>>>(w, wbm, wbs);

  const long M = (long)in_sizes[0] / KD;          // 64*4096 = 262144 rows
  dim3 grid(ND / 128, (unsigned)(M / 128));       // (2, 2048): N-halves adjacent for x L2 reuse
  lns_gemm<<<grid, dim3(256), LDS_BYTES, stream>>>(x, wbm, wbs, bias, out);
}
